// xLSTM_37735582663291
// MI455X (gfx1250) — matrix operand order repaired and checked by reading, not compiled, not run on hardware
//
#include <hip/hip_runtime.h>

// xLSTM (sLSTM -> mLSTM with bilinear) for MI455X / gfx1250, wave32 + WMMA.
//
// All matrix operands are pre-packed into the exact per-lane bf16 WMMA
// fragment layouts (CDNA5 ISA section 7.12.2), so every WMMA in a hot
// loop is fed by b128 loads (global or LDS), not ALU conversion code.
// The bilinear einsum is an outer-product GEMM (M=16,N=256,K=65536) whose
// per-wave A h-fragment is loop-invariant; only a scalar x[m,d] changes per
// K-chunk (v_pk_mul_bf16). The bilinear K-loop is software-pipelined
// (unroll 2, ping-pong B-fragment buffers) so WMMA never waits on the
// just-issued L2 load.
//
// Workspace (bytes):
//   xg1 32MB | h1 8MB | xg2 32MB | u1p 512K | u2p 512K | w1p 1M | w2p 512K |
//   xp 8MB | h1p 4MB | bwp 32MB | counters    (~118.5 MB total)

#define BB 16
#define TT 512
#define DD 512
#define HH 256
#define GG 1024
#define BT (BB * TT)
#define BSCALE 0.1f

typedef __attribute__((ext_vector_type(16))) __bf16 v16bf;
typedef __attribute__((ext_vector_type(8))) float v8f;

union FragB {
  v16bf v;
  unsigned short s[16];
  uint4 q[2];
};

// Hardware f32 -> bf16 (RNE) conversion.
static __device__ __forceinline__ unsigned short f2bf(float f) {
  union { __bf16 b; unsigned short u; } cv;
  cv.b = (__bf16)f;
  return cv.u;
}

static __device__ __forceinline__ v16bf splat16(__bf16 x) {
  v16bf v;
#pragma unroll
  for (int i = 0; i < 16; ++i) v[i] = x;
  return v;
}

static __device__ __forceinline__ float sigm(float x) {
  return 1.0f / (1.0f + __expf(-x));
}

// ---------------------------------------------------------------------------
__global__ void init_counters(unsigned int* c) {
  if (threadIdx.x < 2) c[threadIdx.x] = 0u;
}

// Pack A[Mtiles*16, K] f32 row-major into A-fragment tiles:
// dst[(tile*kcs + kc)*32 + lane][16], lane: m=l%16, runs koff / koff+16.
__global__ void pack_a(const float* __restrict__ A, unsigned short* __restrict__ dst,
                       int K, int Mtiles) {
  int tid = blockIdx.x * blockDim.x + threadIdx.x;
  int kcs = K >> 5;
  if (tid >= Mtiles * kcs * 32) return;
  int lane = tid & 31;
  int kc = (tid >> 5) % kcs;
  int tile = (tid >> 5) / kcs;
  int m = lane & 15;
  int koff = (lane >> 4) * 8;
  const float* src = A + (size_t)(tile * 16 + m) * K + kc * 32 + koff;
  unsigned short* d = dst + (size_t)tid * 16;
#pragma unroll
  for (int i = 0; i < 8; ++i) { d[i] = f2bf(src[i]); d[8 + i] = f2bf(src[16 + i]); }
}

// Pack W[1024, K] f32 row-major (rows = GEMM output columns) into B-fragment
// tiles: dst[(tile*kcs + kc)*32 + lane][16], lane: n=l%16, 8-K runs at k0 / k0+16.
__global__ void pack_b(const float* __restrict__ W, unsigned short* __restrict__ dst,
                       int K) {
  int tid = blockIdx.x * blockDim.x + threadIdx.x;
  int kcs = K >> 5;
  if (tid >= (kcs << 11)) return;   // 64 tiles * 32 lanes * kcs
  int lane = tid & 31;
  int kc = (tid >> 5) % kcs;
  int tile = (tid >> 5) / kcs;
  int n = tile * 16 + (lane & 15);
  int k0 = kc * 32 + ((lane >> 4) * 8);
  const float* src = W + (size_t)n * K + k0;
  unsigned short* d = dst + (size_t)tid * 16;
#pragma unroll
  for (int i = 0; i < 8; ++i) { d[i] = f2bf(src[i]); d[8 + i] = f2bf(src[16 + i]); }
}

// Pack Bw[o,d,k] (256^3) into bf16 B-fragments for the bilinear GEMM:
// dst[wg(16)][chunk(2048)][lane(32)][16], chunk c -> d=c/8, k0=(c%8)*32.
__global__ void pack_bw(const float* __restrict__ Bw, unsigned short* __restrict__ dst) {
  int tid = blockIdx.x * blockDim.x + threadIdx.x;
  if (tid >= 16 * 2048 * 32) return;
  int lane = tid & 31;
  int c = (tid >> 5) & 2047;
  int w = tid >> 16;
  int o = w * 16 + (lane & 15);
  int d = c >> 3;
  int k0 = (c & 7) * 32 + ((lane >> 4) * 8);
  const float* src = Bw + ((size_t)o * 256 + d) * 256 + k0;
  unsigned short* ds = dst + (size_t)tid * 16;
#pragma unroll
  for (int i = 0; i < 8; ++i) { ds[i] = f2bf(src[i]); ds[8 + i] = f2bf(src[16 + i]); }
}

// ---------------------------------------------------------------------------
// Fully packed GEMM: C[M,1024] = A @ W^T + bias, K = KCS*32 compile-time.
// One 16x16 tile per wave; inner loop is 4x b128 loads + 1 wmma.
template <int KCS>
__global__ void __launch_bounds__(256) gemm_packed(
    const unsigned short* __restrict__ ap, const unsigned short* __restrict__ bp,
    const float* __restrict__ bias, float* __restrict__ C, int Mtiles) {
  const int lane = threadIdx.x & 31;
  const int wid = blockIdx.x * (blockDim.x >> 5) + (threadIdx.x >> 5);
  if (wid >= Mtiles * 64) return;
  const int tn = wid & 63, tm = wid >> 6;
  const int m = lane & 15;
  const int hi = lane >> 4;
  const uint4* a4 = (const uint4*)ap + ((size_t)tm * KCS * 32 + lane) * 2;
  const uint4* b4 = (const uint4*)bp + ((size_t)tn * KCS * 32 + lane) * 2;
  const float bv = bias[tn * 16 + m];
  v8f acc = {0.f, 0.f, 0.f, 0.f, 0.f, 0.f, 0.f, 0.f};
#pragma unroll
  for (int kc = 0; kc < KCS; ++kc) {
    FragB af, bf;
    af.q[0] = a4[0]; af.q[1] = a4[1];
    bf.q[0] = b4[0]; bf.q[1] = b4[1];
    a4 += 64; b4 += 64;
    acc = __builtin_amdgcn_wmma_f32_16x16x32_bf16(false, af.v, false, bf.v,
                                                  (short)0, acc, false, false);
  }
  float* crow = C + (size_t)(tm * 16 + hi * 8) * GG + tn * 16 + m;
#pragma unroll
  for (int r = 0; r < 8; ++r) crow[(size_t)r * GG] = acc[r] + bv;
}

// ---------------------------------------------------------------------------
// Agent-scope monotonic barrier across the 16 scan workgroups.
static __device__ __forceinline__ void grid_barrier(unsigned int* bar, unsigned int tgt) {
  __threadfence();
  __syncthreads();
  if (threadIdx.x == 0) {
    __hip_atomic_fetch_add(bar, 1u, __ATOMIC_ACQ_REL, __HIP_MEMORY_SCOPE_AGENT);
    while (__hip_atomic_load(bar, __ATOMIC_ACQUIRE, __HIP_MEMORY_SCOPE_AGENT) < tgt)
      __builtin_amdgcn_s_sleep(1);
  }
  __syncthreads();
}

// ---------------------------------------------------------------------------
// sLSTM scan. 16 WGs; WG w owns h columns [16w,16w+16). 8 waves / WG.
__global__ void __launch_bounds__(256) slstm_scan(
    const float* __restrict__ xg1, const unsigned short* __restrict__ u1p,
    const float* __restrict__ u1b, const float* __restrict__ alpha,
    float* __restrict__ h1, unsigned int* __restrict__ bar) {
  __shared__ __align__(16) unsigned short sh_hb[BB * HH];  // prev h, bf16
  __shared__ float sh_g[4 * BB * 16];                      // gates [q][b][j]
  const int w = blockIdx.x;
  const int tid = threadIdx.x;
  const int lane = tid & 31, wv = tid >> 5;
  const int b_e = tid >> 4, j_e = tid & 15;
  const float alpha_j = alpha[w * 16 + j_e];
  float c_state = 0.f;

  for (int t = 0; t < TT; ++t) {
    for (int i = tid; i < BB * HH; i += 256)
      sh_hb[i] = (t > 0)
          ? f2bf(h1[(((size_t)(i >> 8)) * TT + (t - 1)) * HH + (i & 255)])
          : (unsigned short)0;
    __syncthreads();

    if (wv < 4) {                              // waves 0..3: one gate each
      const int q = wv;
      const int m = lane & 15, hi = lane >> 4, koff = hi * 8;
      v8f acc = {0.f, 0.f, 0.f, 0.f, 0.f, 0.f, 0.f, 0.f};
      if (t > 0) {
        const unsigned short* hrow = sh_hb + m * HH;
        const uint4* b4 = (const uint4*)u1p + ((size_t)(q * 16 + w) * 256 + lane) * 2;
#pragma unroll
        for (int kc = 0; kc < 8; ++kc) {
          FragB af, bf;
          af.q[0] = *(const uint4*)(hrow + kc * 32 + koff);
          af.q[1] = *(const uint4*)(hrow + kc * 32 + koff + 16);
          bf.q[0] = b4[0]; bf.q[1] = b4[1];
          b4 += 64;
          acc = __builtin_amdgcn_wmma_f32_16x16x32_bf16(false, af.v, false, bf.v,
                                                        (short)0, acc, false, false);
        }
      }
      const int gcol = q * 256 + w * 16 + m;
      const float ub = u1b[gcol];
#pragma unroll
      for (int r = 0; r < 8; ++r) {
        int b = r + hi * 8;
        sh_g[(q * BB + b) * 16 + m] =
            acc[r] + xg1[(((size_t)b) * TT + t) * GG + gcol] + ub;
      }
    }
    __syncthreads();

    {
      float gi = sh_g[(0 * BB + b_e) * 16 + j_e];
      float gf = sh_g[(1 * BB + b_e) * 16 + j_e];
      float go = sh_g[(2 * BB + b_e) * 16 + j_e];
      float gg = sh_g[(3 * BB + b_e) * 16 + j_e];
      c_state = alpha_j * (sigm(gf) * c_state + sigm(gi) * tanhf(gg));
      float h = sigm(go) * tanhf(c_state);
      h1[(((size_t)b_e) * TT + t) * HH + w * 16 + j_e] = h;
    }
    grid_barrier(bar, 16u * (unsigned)(t + 1));
  }
}

// ---------------------------------------------------------------------------
// mLSTM scan with bilinear term. 16 WGs; WG w owns o/h columns [16w,16w+16).
// Bilinear K-chunks of wave wv all have k0 = 32*wv, so the h-fragment is
// loop-invariant; per chunk only x[m,d] changes (v_pk_mul_bf16 scaling).
// The 256-chunk loop is unrolled x2 with ping-pong B buffers: loads for
// chunk it+1 are issued before the wmma of chunk it.
__global__ void __launch_bounds__(256) mlstm_scan(
    const float* __restrict__ xg2, const unsigned short* __restrict__ u2p,
    const float* __restrict__ u2b, const unsigned short* __restrict__ bwp,
    const float* __restrict__ bb, const float* __restrict__ x2,
    float* __restrict__ out, unsigned int* __restrict__ bar) {
  __shared__ __align__(16) unsigned short sh_hb[BB * HH];  // prev h, bf16
  __shared__ float sh_x[BB * HH];                          // x_t (= h1[:,t,:])
  __shared__ float sh_g[4 * BB * 16];
  __shared__ float sh_red[8 * 256];                        // bilinear partials
  const int w = blockIdx.x;
  const int tid = threadIdx.x;
  const int lane = tid & 31, wv = tid >> 5;
  const int b_e = tid >> 4, j_e = tid & 15;
  const int m = lane & 15, hi = lane >> 4, koff = hi * 8;
  const float bbj = bb[w * 16 + j_e];
  float c_state = 0.f;

  for (int t = 0; t < TT; ++t) {
    for (int i = tid; i < BB * HH; i += 256) {
      int b = i >> 8, k = i & 255;
      sh_x[i] = x2[(((size_t)b) * TT + t) * HH + k];
      sh_hb[i] = (t > 0) ? f2bf(out[(((size_t)b) * TT + (t - 1)) * HH + k])
                         : (unsigned short)0;
    }
    __syncthreads();

    if (wv < 4) {                              // gate GEMM (waves 0..3)
      const int q = wv;
      v8f acc = {0.f, 0.f, 0.f, 0.f, 0.f, 0.f, 0.f, 0.f};
      if (t > 0) {
        const unsigned short* hrow = sh_hb + m * HH;
        const uint4* b4 = (const uint4*)u2p + ((size_t)(q * 16 + w) * 256 + lane) * 2;
#pragma unroll
        for (int kc = 0; kc < 8; ++kc) {
          FragB af, bf;
          af.q[0] = *(const uint4*)(hrow + kc * 32 + koff);
          af.q[1] = *(const uint4*)(hrow + kc * 32 + koff + 16);
          bf.q[0] = b4[0]; bf.q[1] = b4[1];
          b4 += 64;
          acc = __builtin_amdgcn_wmma_f32_16x16x32_bf16(false, af.v, false, bf.v,
                                                        (short)0, acc, false, false);
        }
      }
      const int gcol = q * 256 + w * 16 + m;
      const float ub = u2b[gcol];
#pragma unroll
      for (int r = 0; r < 8; ++r) {
        int b = r + hi * 8;
        sh_g[(q * BB + b) * 16 + m] =
            acc[r] + xg2[(((size_t)b) * TT + t) * GG + gcol] + ub;
      }
    }

    {  // bilinear: all 8 waves, K = 65536 split by wave
      v8f acc = {0.f, 0.f, 0.f, 0.f, 0.f, 0.f, 0.f, 0.f};
      if (t > 0) {
        const int k0 = wv * 32;
        FragB hf;                              // loop-invariant h fragment
        hf.q[0] = *(const uint4*)(sh_hb + m * HH + k0 + koff);
        hf.q[1] = *(const uint4*)(sh_hb + m * HH + k0 + koff + 16);
        const float* xr = sh_x + m * HH;
        const uint4* b4 = (const uint4*)bwp + ((size_t)(w * 2048 + wv) * 32 + lane) * 2;
        FragB bA, bB;                          // ping-pong B-fragment buffers
        bA.q[0] = b4[0]; bA.q[1] = b4[1];      // chunk it=0
        for (int it = 0; it < 256; it += 2) {
          // issue loads for it+1 before consuming it
          bB.q[0] = b4[512]; bB.q[1] = b4[513];
          if (it + 6 < 256)
            __builtin_prefetch((const void*)(b4 + 512 * 6), 0, 3);
          FragB af;
          af.v = hf.v * splat16((__bf16)xr[it]);      // v_pk_mul_bf16 x8
          acc = __builtin_amdgcn_wmma_f32_16x16x32_bf16(false, af.v, false, bA.v,
                                                        (short)0, acc, false, false);
          // issue loads for it+2 before consuming it+1
          if (it + 2 < 256) { bA.q[0] = b4[1024]; bA.q[1] = b4[1025]; }
          FragB af2;
          af2.v = hf.v * splat16((__bf16)xr[it + 1]);
          acc = __builtin_amdgcn_wmma_f32_16x16x32_bf16(false, af2.v, false, bB.v,
                                                        (short)0, acc, false, false);
          b4 += 1024;                          // two chunks of this wave
        }
      }
#pragma unroll
      for (int r = 0; r < 8; ++r)
        sh_red[wv * 256 + (r + hi * 8) * 16 + m] = acc[r];
    }
    __syncthreads();

    {
      float gi = sh_g[(0 * BB + b_e) * 16 + j_e];
      float gf = sh_g[(1 * BB + b_e) * 16 + j_e];
      float go = sh_g[(2 * BB + b_e) * 16 + j_e];
      float gg = sh_g[(3 * BB + b_e) * 16 + j_e];
      float mraw = 0.f;
#pragma unroll
      for (int v = 0; v < 8; ++v) mraw += sh_red[v * 256 + tid];
      float mm = tanhf(mraw + bbj);
      c_state = sigm(gf) * c_state + sigm(gi) * tanhf(gg) + BSCALE * mm;
      float h = sigm(go) * tanhf(c_state);
      out[(((size_t)b_e) * TT + t) * HH + w * 16 + j_e] = h;
    }
    grid_barrier(bar, 16u * (unsigned)(t + 1));
  }
}

// ---------------------------------------------------------------------------
extern "C" void kernel_launch(void* const* d_in, const int* in_sizes, int n_in,
                              void* d_out, int out_size, void* d_ws, size_t ws_size,
                              hipStream_t stream) {
  (void)in_sizes; (void)n_in; (void)out_size; (void)ws_size;
  const float* x     = (const float*)d_in[0];
  const float* W1w   = (const float*)d_in[1];
  const float* W1b   = (const float*)d_in[2];
  const float* U1w   = (const float*)d_in[3];
  const float* U1b   = (const float*)d_in[4];
  const float* alpha = (const float*)d_in[5];
  const float* W2w   = (const float*)d_in[6];
  const float* W2b   = (const float*)d_in[7];
  const float* U2w   = (const float*)d_in[8];
  const float* U2b   = (const float*)d_in[9];
  const float* Bw    = (const float*)d_in[10];
  const float* Bb    = (const float*)d_in[11];
  float* out = (float*)d_out;

  char* ws = (char*)d_ws;
  float* xg1 = (float*)ws;                             // 8,388,608 f
  float* h1  = xg1 + (size_t)BT * GG;                  // 2,097,152 f
  float* xg2 = h1 + (size_t)BT * HH;                   // 8,388,608 f
  unsigned short* u1p = (unsigned short*)(ws + 75497472ull);   // 512K
  unsigned short* u2p = (unsigned short*)(ws + 76021760ull);   // 512K
  unsigned short* w1p = (unsigned short*)(ws + 76546048ull);   // 1M
  unsigned short* w2p = (unsigned short*)(ws + 77594624ull);   // 512K
  unsigned short* xp  = (unsigned short*)(ws + 78118912ull);   // 8M
  unsigned short* h1p = (unsigned short*)(ws + 86507520ull);   // 4M
  unsigned short* bwp = (unsigned short*)(ws + 90701824ull);   // 32M
  unsigned int* bar   = (unsigned int*)(ws + 124256256ull);

  init_counters<<<1, 32, 0, stream>>>(bar);
  pack_b<<<128, 256, 0, stream>>>(W1w, w1p, DD);
  pack_b<<<64, 256, 0, stream>>>(W2w, w2p, HH);
  pack_b<<<64, 256, 0, stream>>>(U1w, u1p, HH);
  pack_b<<<64, 256, 0, stream>>>(U2w, u2p, HH);
  pack_bw<<<4096, 256, 0, stream>>>(Bw, bwp);
  pack_a<<<1024, 256, 0, stream>>>(x, xp, DD, BT / 16);

  gemm_packed<16><<<4096, 256, 0, stream>>>(xp, w1p, W1b, xg1, BT / 16);
  slstm_scan<<<16, 256, 0, stream>>>(xg1, u1p, U1b, alpha, h1, bar + 0);
  pack_a<<<512, 256, 0, stream>>>(h1, h1p, HH, BT / 16);
  gemm_packed<8><<<4096, 256, 0, stream>>>(h1p, w2p, W2b, xg2, BT / 16);
  mlstm_scan<<<16, 256, 0, stream>>>(xg2, u2p, U2b, bwp, Bb, h1, out, bar + 1);
}
